// EvoFormer_89661737271785
// MI455X (gfx1250) — matrix operand order repaired and checked by reading, not compiled, not run on hardware
//
#include <hip/hip_runtime.h>
#include <math.h>

// ---------------- types for WMMA ----------------
typedef __bf16 bf16;
typedef bf16  v16bf __attribute__((ext_vector_type(16)));
typedef float v8f   __attribute__((ext_vector_type(8)));
typedef unsigned int v8u __attribute__((ext_vector_type(8)));

// ---------------- problem dims ----------------
static constexpr int C_M = 256, S_ = 128, L_ = 192;
static constexpr int C_Z = 128, C_HID = 32, H_MSA = 8, H_TRI = 4, C_TRI = 128;
static constexpr int PM = S_ * L_;          // 24576 msa positions
static constexpr int PP = L_ * L_;          // 36864 pair positions
static constexpr long long NM = (long long)C_M * PM;   // 6291456
static constexpr long long NP = (long long)C_Z * PP;   // 4718592
#define EPS_LN 1e-5f

// ---------------- fp32 pair -> packed bf16 dword (round-half-up) ----------------
__device__ __forceinline__ unsigned int packbf(float a, float b) {
  unsigned int ra = __float_as_uint(a) + 0x8000u;
  unsigned int rb = __float_as_uint(b) + 0x8000u;
  return (rb & 0xFFFF0000u) | (ra >> 16);
}

// =====================================================================
// Generalized batched strided GEMM via v_wmma_f32_16x16x32_bf16.
//   D[b][m][n] (+)= ep( alpha * sum_k A[b](m,k)*B[b](k,n) + bias[m] )
//   operand base for batch b: (b/div)*hi + (b%div)*lo; then m*Sm + k*Sk etc.
// Block = 128 threads (4 wave32s); block tile M32 x N64; wave w owns the
// n-sub-tile [16w,16w+16) and computes two stacked 16x16 WMMAs (m and m+16)
// reusing one B fragment. LDS tiles are stored k-packed (2 bf16 per dword)
// so every fragment load is a single ds_load_b32.
// Edge handling is branch-free: out-of-range m/n staging addresses are
// CLAMPED (garbage only reaches D rows/cols that are never stored).
// REQUIREMENT: K is a multiple of 32 (true for every call below).
// =====================================================================
struct Gemm {
  const float* A; const float* Bm; float* D; const float* bias;
  int M, N, K, batch;
  long long Adiv, Ahi, Alo, Am, Ak;
  long long Bdiv, Bhi, Blo, Bk, Bn;
  long long Ddiv, Dhi, Dlo, Dm, Dn;
  float alpha; int ep; int accum;   // ep: 0 none, 1 relu, 2 sigmoid
};

template <bool AK1>
__global__ __launch_bounds__(128) void gemm_wmma(Gemm g) {
  __shared__ unsigned int As[32][17];  // rows m(32) x kp(16), kp = k/2 packed
  __shared__ unsigned int Bs[16][66];  // kp(16) x n(64)
  const int tid  = threadIdx.x;
  const int lane = tid & 31;
  const int wave = tid >> 5;
  const int m0 = blockIdx.x * 32;
  const int n0 = blockIdx.y * 64;
  const long long bz = blockIdx.z;
  const long long baseA = (bz / g.Adiv) * g.Ahi + (bz % g.Adiv) * g.Alo;
  const long long baseB = (bz / g.Bdiv) * g.Bhi + (bz % g.Bdiv) * g.Blo;
  const long long baseD = (bz / g.Ddiv) * g.Dhi + (bz % g.Ddiv) * g.Dlo;

  v8f acc0 = {}, acc1 = {};
  const int ml = lane & 15;
  const int hi = lane >> 4;
  const int nn = wave * 16 + ml;

  // clamped staging coordinates (branch-free edge handling)
  const int mclamp = g.M - 1, nclamp = g.N - 1;

  for (int kb = 0; kb < g.K; kb += 32) {
    // stage A 32x32 -> 512 packed dwords, 4 per thread
#pragma unroll
    for (int t = 0; t < 4; ++t) {
      int idx = t * 128 + tid;
      int mi = idx >> 4, kp = idx & 15;
      int gm = m0 + mi; gm = gm < mclamp ? gm : mclamp;
      int gk = kb + 2 * kp;
      long long a = baseA + (long long)gm * g.Am + (long long)gk * g.Ak;
      float v0, v1;
      if (AK1) {
        float2 p = *(const float2*)(g.A + a);   // aligned: gk even, Am even
        v0 = p.x; v1 = p.y;
      } else {
        v0 = g.A[a]; v1 = g.A[a + g.Ak];
      }
      As[mi][kp] = packbf(v0, v1);
    }
    // stage B 32x64 -> 1024 packed dwords, 8 per thread (n-coalesced)
#pragma unroll
    for (int t = 0; t < 8; ++t) {
      int idx = t * 128 + tid;
      int kp = idx >> 6, bn = idx & 63;
      int gn = n0 + bn; gn = gn < nclamp ? gn : nclamp;
      int gk = kb + 2 * kp;
      long long a = baseB + (long long)gk * g.Bk + (long long)gn * g.Bn;
      Bs[kp][bn] = packbf(g.Bm[a], g.Bm[a + g.Bk]);
    }
    __syncthreads();

    // fragments: A lane(m=ml)+VGPR(u) -> kp = u+4*hi (u<4) / u+4+4*hi (u>=4)
    //            B lane(n)+VGPR(u)    -> same kp as A
    v8u au0, au1, bu;
#pragma unroll
    for (int u = 0; u < 8; ++u) {
      int kp = (u < 4) ? (u + 4 * hi) : (u + 4 + 4 * hi);
      au0[u] = As[ml][kp];
      au1[u] = As[16 + ml][kp];
      bu[u]  = Bs[kp][nn];
    }
    v16bf af0 = __builtin_bit_cast(v16bf, au0);
    v16bf af1 = __builtin_bit_cast(v16bf, au1);
    v16bf bfr = __builtin_bit_cast(v16bf, bu);
    acc0 = __builtin_amdgcn_wmma_f32_16x16x32_bf16(false, af0, false, bfr,
                                                   (short)0, acc0, false, false);
    acc1 = __builtin_amdgcn_wmma_f32_16x16x32_bf16(false, af1, false, bfr,
                                                   (short)0, acc1, false, false);
    __syncthreads();
  }

  // store: lane -> n = n0+16*wave+ml; VGPR r -> m = r + 8*hi (+16 for acc1)
  const int n = n0 + wave * 16 + ml;
  if (n < g.N) {
#pragma unroll
    for (int halfm = 0; halfm < 2; ++halfm) {
      v8f acc = halfm ? acc1 : acc0;
      int mb = m0 + 16 * halfm;
#pragma unroll
      for (int r = 0; r < 8; ++r) {
        int m = mb + r + 8 * hi;
        if (m < g.M) {
          float v = g.alpha * acc[r];
          if (g.bias) v += g.bias[m];
          if (g.ep == 1)      v = v > 0.f ? v : 0.f;
          else if (g.ep == 2) v = 1.f / (1.f + __expf(-v));
          long long o = baseD + (long long)m * g.Dm + (long long)n * g.Dn;
          if (g.accum) g.D[o] += v; else g.D[o] = v;
        }
      }
    }
  }
}

// =====================================================================
// Global layernorm (normalize jointly over whole tensor, B=1):
// two-pass deterministic stats + elementwise apply with full-size affine.
// =====================================================================
__global__ void stat_part(const float* __restrict__ x, long long n, float* part) {
  __shared__ float ss[256], sq[256];
  float a = 0.f, b = 0.f;
  for (long long i = (long long)blockIdx.x * 256 + threadIdx.x; i < n;
       i += 256ll * gridDim.x) {
    float v = x[i]; a += v; b += v * v;
  }
  ss[threadIdx.x] = a; sq[threadIdx.x] = b; __syncthreads();
  for (int s = 128; s > 0; s >>= 1) {
    if (threadIdx.x < s) { ss[threadIdx.x] += ss[threadIdx.x + s];
                           sq[threadIdx.x] += sq[threadIdx.x + s]; }
    __syncthreads();
  }
  if (threadIdx.x == 0) { part[2 * blockIdx.x] = ss[0]; part[2 * blockIdx.x + 1] = sq[0]; }
}

__global__ void stat_fin(const float* __restrict__ part, int nblk, long long n, float* stat) {
  __shared__ float ss[256], sq[256];
  float a = 0.f, b = 0.f;
  for (int i = threadIdx.x; i < nblk; i += 256) { a += part[2 * i]; b += part[2 * i + 1]; }
  ss[threadIdx.x] = a; sq[threadIdx.x] = b; __syncthreads();
  for (int s = 128; s > 0; s >>= 1) {
    if (threadIdx.x < s) { ss[threadIdx.x] += ss[threadIdx.x + s];
                           sq[threadIdx.x] += sq[threadIdx.x + s]; }
    __syncthreads();
  }
  if (threadIdx.x == 0) {
    float mean = ss[0] / (float)n;
    float var  = sq[0] / (float)n - mean * mean;
    stat[0] = mean; stat[1] = rsqrtf(var + EPS_LN);
  }
}

__global__ void ln_apply(float* __restrict__ y, const float* __restrict__ x,
                         const float* __restrict__ w, const float* __restrict__ b,
                         const float* __restrict__ stat, long long n) {
  const float m = stat[0], r = stat[1];
  for (long long i = (long long)blockIdx.x * 256 + threadIdx.x; i < n;
       i += 256ll * gridDim.x)
    y[i] = (x[i] - m) * r * w[i] + b[i];
}

// =====================================================================
// Softmax over rows of [batch][M][N] (row-major), optional bias:
//   bias[(b/bdiv)*bbs + m*bms + n*bns]
// =====================================================================
struct Smax { float* s; const float* bias; int M, N; long long bdiv, bbs, bms, bns; };

__global__ void softmax_rows(Smax p) {
  const long long row = blockIdx.x;
  const long long b = row / p.M;
  const int m = (int)(row % p.M);
  float* ptr = p.s + (b * (long long)p.M + m) * p.N;
  __shared__ float red[64];
  const int tid = threadIdx.x;
  float vals[3];
  float mx = -INFINITY;
#pragma unroll
  for (int t = 0; t < 3; ++t) {
    int n = tid + 64 * t; float v = -INFINITY;
    if (n < p.N) {
      v = ptr[n];
      if (p.bias)
        v += p.bias[(b / p.bdiv) * p.bbs + (long long)m * p.bms + (long long)n * p.bns];
    }
    vals[t] = v; mx = fmaxf(mx, v);
  }
  red[tid] = mx; __syncthreads();
  for (int s = 32; s > 0; s >>= 1) { if (tid < s) red[tid] = fmaxf(red[tid], red[tid + s]); __syncthreads(); }
  mx = red[0]; __syncthreads();
  float sum = 0.f;
#pragma unroll
  for (int t = 0; t < 3; ++t) {
    int n = tid + 64 * t;
    if (n < p.N) { vals[t] = __expf(vals[t] - mx); sum += vals[t]; }
  }
  red[tid] = sum; __syncthreads();
  for (int s = 32; s > 0; s >>= 1) { if (tid < s) red[tid] += red[tid + s]; __syncthreads(); }
  const float inv = 1.f / red[0];
#pragma unroll
  for (int t = 0; t < 3; ++t) { int n = tid + 64 * t; if (n < p.N) ptr[n] = vals[t] * inv; }
}

// ---------------- elementwise ----------------
__global__ void ew_mul(float* d, const float* s, long long n) {
  for (long long i = (long long)blockIdx.x * 256 + threadIdx.x; i < n; i += 256ll * gridDim.x)
    d[i] *= s[i];
}
__global__ void ew_gadd(float* d, const float* g, const float* x, long long n) {
  for (long long i = (long long)blockIdx.x * 256 + threadIdx.x; i < n; i += 256ll * gridDim.x)
    d[i] += g[i] * x[i];
}
// o is in merged (c*H+h) channel layout; g in standard (h*C+c) layout
__global__ void gate_mul_merged(float* o, const float* g, int H, int C, long long P, long long n) {
  for (long long i = (long long)blockIdx.x * 256 + threadIdx.x; i < n; i += 256ll * gridDim.x) {
    long long ch = i / P, pos = i % P;
    int c = (int)(ch / H), h = (int)(ch % H);
    o[i] *= g[((long long)h * C + c) * P + pos];
  }
}

// =====================================================================
// host-side launch helpers
// =====================================================================
static void gemm_launch(hipStream_t st, const float* A, const float* Bm, float* D,
                        const float* bias, int M, int N, int K, int batch,
                        long long Adiv, long long Ahi, long long Alo, long long Am, long long Ak,
                        long long Bdiv, long long Bhi, long long Blo, long long Bk, long long Bn,
                        long long Ddiv, long long Dhi, long long Dlo, long long Dm, long long Dn,
                        float alpha, int ep, int accum) {
  Gemm g{A, Bm, D, bias, M, N, K, batch,
         Adiv, Ahi, Alo, Am, Ak, Bdiv, Bhi, Blo, Bk, Bn, Ddiv, Dhi, Dlo, Dm, Dn,
         alpha, ep, accum};
  dim3 grid((M + 31) / 32, (N + 63) / 64, batch);
  if (Ak == 1) gemm_wmma<true ><<<grid, dim3(128), 0, st>>>(g);
  else         gemm_wmma<false><<<grid, dim3(128), 0, st>>>(g);
}

// plain conv1x1: D[Cout,P] = W[Cout,Cin] x X[Cin,P]
static void conv(hipStream_t st, const float* W, const float* X, float* D, const float* bias,
                 int Cout, int Cin, int P, int ep, int accum, float alpha = 1.f) {
  gemm_launch(st, W, X, D, bias, Cout, P, Cin, 1,
              1, 0, 0, (long long)Cin, 1,
              1, 0, 0, (long long)P, 1,
              1, 0, 0, (long long)P, 1, alpha, ep, accum);
}

static void lnorm(hipStream_t st, float* dst, const float* x, const float* w, const float* b,
                  long long n, float* wpart, float* wstat) {
  stat_part<<<1024, 256, 0, st>>>(x, n, wpart);
  stat_fin<<<1, 256, 0, st>>>(wpart, 1024, n, wstat);
  ln_apply<<<4096, 256, 0, st>>>(dst, x, w, b, wstat, n);
}

struct TriMultP {
  const float *ln1_b, *ln1_w, *ln2_b, *ln2_w, *wa1_b, *wa1_w, *wa2_b, *wa2_w,
              *wb1_b, *wb1_w, *wb2_b, *wb2_w, *wg_b, *wg_w, *wrep_b, *wrep_w;
};
struct TriAttnP {
  const float *ln1_b, *ln1_w, *wb, *wg_b, *wg_w, *wk, *wq, *wrep_b, *wrep_w, *wv;
};

static void tri_mult_mod(hipStream_t st, float* pair, const TriMultP& p, bool outgoing,
                         float* bM, float* bQ, float* bK, float* bV, float* bG, float* bO,
                         float* wpart, float* wstat) {
  lnorm(st, bM, pair, p.ln1_w, p.ln1_b, NP, wpart, wstat);
  conv(st, p.wa1_w, bM, bQ, p.wa1_b, C_TRI, C_Z, PP, 2, 0);   // sigmoid branch
  conv(st, p.wa2_w, bM, bK, p.wa2_b, C_TRI, C_Z, PP, 0, 0);
  ew_mul<<<4096, 256, 0, st>>>(bQ, bK, NP);                   // a -> bQ
  conv(st, p.wb1_w, bM, bK, p.wb1_b, C_TRI, C_Z, PP, 2, 0);
  conv(st, p.wb2_w, bM, bV, p.wb2_b, C_TRI, C_Z, PP, 0, 0);
  ew_mul<<<4096, 256, 0, st>>>(bK, bV, NP);                   // b -> bK
  conv(st, p.wg_w, bM, bG, p.wg_b, C_Z, C_Z, PP, 2, 0);       // g -> bG
  // triangle einsum, batch over channel c (128): M=N=K=L
  if (outgoing)   // D[i,j] = sum_k a[c,i,k]*b[c,j,k]
    gemm_launch(st, bQ, bK, bO, nullptr, L_, L_, L_, C_TRI,
                1, (long long)PP, 0, (long long)L_, 1,
                1, (long long)PP, 0, 1, (long long)L_,
                1, (long long)PP, 0, (long long)L_, 1, 1.f, 0, 0);
  else            // D[i,j] = sum_k a[c,k,i]*b[c,k,j]
    gemm_launch(st, bQ, bK, bO, nullptr, L_, L_, L_, C_TRI,
                1, (long long)PP, 0, 1, (long long)L_,
                1, (long long)PP, 0, (long long)L_, 1,
                1, (long long)PP, 0, (long long)L_, 1, 1.f, 0, 0);
  lnorm(st, bV, bO, p.ln2_w, p.ln2_b, NP, wpart, wstat);      // ln2 -> bV
  conv(st, p.wrep_w, bV, bO, p.wrep_b, C_Z, C_TRI, PP, 0, 0); // rep -> bO
  ew_gadd<<<4096, 256, 0, st>>>(pair, bG, bO, NP);            // pair += g*rep
}

static void tri_attn_mod(hipStream_t st, float* pair, const TriAttnP& p, bool start,
                         float* bM, float* bQ, float* bK, float* bV, float* bG, float* bO,
                         float* bBig, float* bBias, float* wpart, float* wstat) {
  const float scale = 1.f / sqrtf((float)C_HID);
  lnorm(st, bM, pair, p.ln1_w, p.ln1_b, NP, wpart, wstat);
  conv(st, p.wq, bM, bQ, nullptr, C_Z, C_Z, PP, 0, 0);
  conv(st, p.wk, bM, bK, nullptr, C_Z, C_Z, PP, 0, 0);
  conv(st, p.wv, bM, bV, nullptr, C_Z, C_Z, PP, 0, 0);
  conv(st, p.wg_w, bM, bG, p.wg_b, C_Z, C_Z, PP, 2, 0);
  conv(st, p.wb, bM, bBias, nullptr, H_TRI, C_Z, PP, 0, 0);   // bmat [H,L,L]
  const int batch = H_TRI * L_;
  if (start) {
    // scores[i][j,k] = sum_c q[c,i,j]*k[c,i,k]; batch b=(h,i)
    gemm_launch(st, bQ, bK, bBig, nullptr, L_, L_, C_HID, batch,
                L_, 32ll * PP, (long long)L_, 1, (long long)PP,
                L_, 32ll * PP, (long long)L_, (long long)PP, 1,
                1, (long long)PP, 0, (long long)L_, 1, scale, 0, 0);
    Smax sp{bBig, bBias, L_, L_, (long long)L_, (long long)PP, (long long)L_, 1};
    softmax_rows<<<(unsigned)(batch * L_), 64, 0, st>>>(sp);
    // o[c,i,j] = sum_k a[i,j,k]*v[c,i,k]; write merged (c*H+h)
    gemm_launch(st, bV, bBig, bO, nullptr, C_HID, L_, L_, batch,
                L_, 32ll * PP, (long long)L_, (long long)PP, 1,
                1, (long long)PP, 0, 1, (long long)L_,
                L_, (long long)PP, (long long)L_, (long long)H_TRI * PP, 1, 1.f, 0, 0);
  } else {
    // scores[i,k] per (h,j): sum_c q[c,i,j]*k[c,k,j]; bias bmat[h,k,i]
    gemm_launch(st, bQ, bK, bBig, nullptr, L_, L_, C_HID, batch,
                L_, 32ll * PP, 1, (long long)L_, (long long)PP,
                L_, 32ll * PP, 1, (long long)PP, (long long)L_,
                1, (long long)PP, 0, (long long)L_, 1, scale, 0, 0);
    Smax sp{bBig, bBias, L_, L_, (long long)L_, (long long)PP, 1, (long long)L_};
    softmax_rows<<<(unsigned)(batch * L_), 64, 0, st>>>(sp);
    // o[c,i,j] = sum_k a[i,k]*v[c,k,j]; write merged
    gemm_launch(st, bV, bBig, bO, nullptr, C_HID, L_, L_, batch,
                L_, 32ll * PP, 1, (long long)PP, (long long)L_,
                1, (long long)PP, 0, 1, (long long)L_,
                L_, (long long)PP, 1, (long long)H_TRI * PP, (long long)L_, 1.f, 0, 0);
  }
  gate_mul_merged<<<4096, 256, 0, st>>>(bO, bG, H_TRI, C_HID, (long long)PP, NP);
  conv(st, p.wrep_w, bO, pair, p.wrep_b, C_Z, H_TRI * C_HID, PP, 0, 1);
}

// =====================================================================
// top-level orchestration
// =====================================================================
extern "C" void kernel_launch(void* const* d_in, const int* in_sizes, int n_in,
                              void* d_out, int out_size, void* d_ws, size_t ws_size,
                              hipStream_t stream) {
  (void)in_sizes; (void)n_in; (void)out_size; (void)ws_size;
  const float* msa_in  = (const float*)d_in[0];
  const float* pair_in = (const float*)d_in[1];
  int x = 2;
#define NEXT ((const float*)d_in[x++])
  // a1 (sorted keys)
  const float* a1_ln1_b = NEXT; const float* a1_ln1_w = NEXT;
  const float* a1_ln2_b = NEXT; const float* a1_ln2_w = NEXT;
  const float* a1_wb = NEXT; const float* a1_wg_b = NEXT; const float* a1_wg_w = NEXT;
  const float* a1_wk = NEXT; const float* a1_wq = NEXT;
  const float* a1_wrep_b = NEXT; const float* a1_wrep_w = NEXT; const float* a1_wv = NEXT;
  // a2
  const float* a2_ln1_b = NEXT; const float* a2_ln1_w = NEXT;
  const float* a2_wg_b = NEXT; const float* a2_wg_w = NEXT;
  const float* a2_wk = NEXT; const float* a2_wq = NEXT;
  const float* a2_wrep_b = NEXT; const float* a2_wrep_w = NEXT; const float* a2_wv = NEXT;
  // opm
  const float* opm_ln_b = NEXT; const float* opm_ln_w = NEXT;
  const float* opm_wa_b = NEXT; const float* opm_wa_w = NEXT;
  const float* opm_wb_b = NEXT; const float* opm_wb_w = NEXT;
  const float* opm_wz_b = NEXT; const float* opm_wz_w = NEXT;
  // trans1, trans2
  const float* t1_l1 = NEXT; const float* t1_l2 = NEXT;
  const float* t1_ln_b = NEXT; const float* t1_ln_w = NEXT;
  const float* t2_l1 = NEXT; const float* t2_l2 = NEXT;
  const float* t2_ln_b = NEXT; const float* t2_ln_w = NEXT;
  // tri1, tri2 (tri_mult)
  TriMultP tm[2];
  for (int t = 0; t < 2; ++t) {
    tm[t].ln1_b = NEXT; tm[t].ln1_w = NEXT; tm[t].ln2_b = NEXT; tm[t].ln2_w = NEXT;
    tm[t].wa1_b = NEXT; tm[t].wa1_w = NEXT; tm[t].wa2_b = NEXT; tm[t].wa2_w = NEXT;
    tm[t].wb1_b = NEXT; tm[t].wb1_w = NEXT; tm[t].wb2_b = NEXT; tm[t].wb2_w = NEXT;
    tm[t].wg_b  = NEXT; tm[t].wg_w  = NEXT; tm[t].wrep_b = NEXT; tm[t].wrep_w = NEXT;
  }
  // tri3, tri4 (tri_attn)
  TriAttnP ta[2];
  for (int t = 0; t < 2; ++t) {
    ta[t].ln1_b = NEXT; ta[t].ln1_w = NEXT; ta[t].wb = NEXT;
    ta[t].wg_b = NEXT; ta[t].wg_w = NEXT; ta[t].wk = NEXT; ta[t].wq = NEXT;
    ta[t].wrep_b = NEXT; ta[t].wrep_w = NEXT; ta[t].wv = NEXT;
  }
#undef NEXT

  // ---- workspace arena (floats) ----
  float* ws = (float*)d_ws;
  float* wpart = ws;             // 2048 floats
  float* wstat = ws + 4096;      // 2 floats
  long long off = 8192;
  float* bM = ws + off; off += NM;
  float* bQ = ws + off; off += NM;
  float* bK = ws + off; off += NM;
  float* bV = ws + off; off += NM;
  float* bG = ws + off; off += NM;
  float* bO = ws + off; off += NM;
  float* bT = ws + off; off += NM;
  float* bBig  = ws + off; off += 1024ll * PP;          // scores / hidden / opm
  float* bBias = ws + off; off += (long long)H_MSA * PP;

  float* msa  = (float*)d_out;
  float* pair = msa + NM;
  hipMemcpyAsync(msa, msa_in, NM * sizeof(float), hipMemcpyDeviceToDevice, stream);
  hipMemcpyAsync(pair, pair_in, NP * sizeof(float), hipMemcpyDeviceToDevice, stream);

  const float scale = 1.f / sqrtf((float)C_HID);
  const long long SL = (long long)S_ * L_;

  // ================= row attention (a1) =================
  lnorm(stream, bM, msa, a1_ln1_w, a1_ln1_b, NM, wpart, wstat);
  conv(stream, a1_wq, bM, bQ, nullptr, C_M, C_M, PM, 0, 0);
  conv(stream, a1_wk, bM, bK, nullptr, C_M, C_M, PM, 0, 0);
  conv(stream, a1_wv, bM, bV, nullptr, C_M, C_M, PM, 0, 0);
  conv(stream, a1_wg_w, bM, bG, a1_wg_b, C_M, C_M, PM, 2, 0);
  lnorm(stream, bT, pair, a1_ln2_w, a1_ln2_b, NP, wpart, wstat);
  conv(stream, a1_wb, bT, bBias, nullptr, H_MSA, C_Z, PP, 0, 0);   // pair bias [H,L,L]
  // scores[i,j] per (h,s): sum_c q[c,s,i]*k[c,s,j]
  gemm_launch(stream, bQ, bK, bBig, nullptr, L_, L_, C_HID, H_MSA * S_,
              S_, 32ll * SL, (long long)L_, 1, SL,
              S_, 32ll * SL, (long long)L_, SL, 1,
              1, (long long)PP, 0, (long long)L_, 1, scale, 0, 0);
  {
    Smax sp{bBig, bBias, L_, L_, (long long)S_, (long long)PP, (long long)L_, 1};
    softmax_rows<<<(unsigned)(H_MSA * S_ * L_), 64, 0, stream>>>(sp);
  }
  // o[c,i] per (h,s) = sum_j v[c,j]*a[i,j]; write merged (c*H+h)
  gemm_launch(stream, bV, bBig, bO, nullptr, C_HID, L_, L_, H_MSA * S_,
              S_, 32ll * SL, (long long)L_, SL, 1,
              1, (long long)PP, 0, 1, (long long)L_,
              S_, SL, (long long)L_, (long long)H_MSA * SL, 1, 1.f, 0, 0);
  gate_mul_merged<<<4096, 256, 0, stream>>>(bO, bG, H_MSA, C_HID, (long long)PM, NM);
  conv(stream, a1_wrep_w, bO, msa, a1_wrep_b, C_M, C_M, PM, 0, 1);

  // ================= column attention (a2) =================
  lnorm(stream, bM, msa, a2_ln1_w, a2_ln1_b, NM, wpart, wstat);
  conv(stream, a2_wq, bM, bQ, nullptr, C_M, C_M, PM, 0, 0);
  conv(stream, a2_wk, bM, bK, nullptr, C_M, C_M, PM, 0, 0);
  conv(stream, a2_wv, bM, bV, nullptr, C_M, C_M, PM, 0, 0);
  conv(stream, a2_wg_w, bM, bG, a2_wg_b, C_M, C_M, PM, 2, 0);
  // scores[s,t] per (h,i): sum_c q[c,s,i]*k[c,t,i]
  gemm_launch(stream, bQ, bK, bBig, nullptr, S_, S_, C_HID, H_MSA * L_,
              L_, 32ll * SL, 1, (long long)L_, SL,
              L_, 32ll * SL, 1, SL, (long long)L_,
              1, (long long)S_ * S_, 0, (long long)S_, 1, scale, 0, 0);
  {
    Smax sp{bBig, nullptr, S_, S_, 1, 0, 0, 0};
    softmax_rows<<<(unsigned)(H_MSA * L_ * S_), 64, 0, stream>>>(sp);
  }
  // o[c,s] per (h,i) = sum_t v[c,t]*a[s,t]; write merged
  gemm_launch(stream, bV, bBig, bO, nullptr, C_HID, S_, S_, H_MSA * L_,
              L_, 32ll * SL, 1, SL, (long long)L_,
              1, (long long)S_ * S_, 0, 1, (long long)S_,
              L_, SL, 1, (long long)H_MSA * SL, (long long)L_, 1.f, 0, 0);
  gate_mul_merged<<<4096, 256, 0, stream>>>(bO, bG, H_MSA, C_HID, (long long)PM, NM);
  conv(stream, a2_wrep_w, bO, msa, a2_wrep_b, C_M, C_M, PM, 0, 1);

  // ================= msa transition =================
  lnorm(stream, bM, msa, t1_ln_w, t1_ln_b, NM, wpart, wstat);
  conv(stream, t1_l1, bM, bBig, nullptr, 4 * C_M, C_M, PM, 1, 0);   // relu
  conv(stream, t1_l2, bBig, msa, nullptr, C_M, 4 * C_M, PM, 0, 1);

  // ================= outer product mean =================
  lnorm(stream, bM, msa, opm_ln_w, opm_ln_b, NM, wpart, wstat);
  conv(stream, opm_wa_w, bM, bQ, opm_wa_b, C_HID, C_M, PM, 0, 0);
  conv(stream, opm_wb_w, bM, bK, opm_wb_b, C_HID, C_M, PM, 0, 0);
  // o[(c,v)] per (i,j) = (1/S) sum_s a[c,s,i]*b[v,s,j]; batch = i*L+j
  gemm_launch(stream, bQ, bK, bBig, nullptr, C_HID, C_HID, S_, PP,
              L_, 1, 0, SL, (long long)L_,
              L_, 0, 1, (long long)L_, SL,
              1, 1, 0, 32ll * PP, (long long)PP, 1.f / S_, 0, 0);
  conv(stream, opm_wz_w, bBig, pair, opm_wz_b, C_Z, C_HID * C_HID, PP, 0, 1);

  // ================= triangle updates =================
  tri_mult_mod(stream, pair, tm[0], true,  bM, bQ, bK, bV, bG, bO, wpart, wstat);
  tri_mult_mod(stream, pair, tm[1], false, bM, bQ, bK, bV, bG, bO, wpart, wstat);
  tri_attn_mod(stream, pair, ta[0], true,  bM, bQ, bK, bV, bG, bO, bBig, bBias, wpart, wstat);
  tri_attn_mod(stream, pair, ta[1], false, bM, bQ, bK, bV, bG, bO, bBig, bBias, wpart, wstat);

  // ================= pair transition =================
  lnorm(stream, bM, pair, t2_ln_w, t2_ln_b, NP, wpart, wstat);
  conv(stream, t2_l1, bM, bBig, nullptr, 4 * C_Z, C_Z, PP, 1, 0);
  conv(stream, t2_l2, bBig, pair, nullptr, C_Z, 4 * C_Z, PP, 0, 1);
}
